// drum_gen_model_45681272160309
// MI455X (gfx1250) — compile-verified, matrix operand order checked by reading, not run on hardware
//
#include <hip/hip_runtime.h>
#include <hip/hip_bf16.h>
#include <math.h>

typedef _Float16 half_t;
typedef __attribute__((ext_vector_type(16))) _Float16 v16h;
typedef __attribute__((ext_vector_type(8)))  _Float16 v8h;
typedef __attribute__((ext_vector_type(8)))  float    v8f;

#define BB 1024   // batch
#define TT 64     // timesteps

__device__ __forceinline__ float sigf(float v) { return 1.0f / (1.0f + __expf(-v)); }

// ---------------------------------------------------------------------------
// WMMA GEMM: C[M,N] = act(A[M,K](f16,lda) x Bt[N,K](f16) + bias[N])
// Requirements (guaranteed by host): N % 64 == 0, K % 32 == 0, M % 16 == 0,
// grid covers exactly (no bounds checks -> no divergence around WMMA).
// Each wave computes a 16(M) x 64(N) strip = 4 WMMA tiles, reusing the A frag.
// Per ISA 16-bit A/B layout: lane L holds row (L&15); K-halves
// [kb + hi*8 .. +7] and [kb + 16 + hi*8 .. +7] with hi = L>>4.
// ---------------------------------------------------------------------------
__global__ __launch_bounds__(256)
void gemm16_kernel(const half_t* __restrict__ A, int lda,
                   const half_t* __restrict__ Bt,
                   float* __restrict__ C, int ldc,
                   half_t* __restrict__ C16, int ldc16,
                   const float* __restrict__ bias,
                   int K, int act)
{
    const int lane  = threadIdx.x & 31;
    const int wave  = threadIdx.x >> 5;
    const int strip = blockIdx.x * (blockDim.x >> 5) + wave;
    const int n0    = strip * 64;
    const int m0    = blockIdx.y * 16;
    const int hi    = lane >> 4;
    const int r15   = lane & 15;

    const v8f vzero = {0.f,0.f,0.f,0.f,0.f,0.f,0.f,0.f};
    v8f acc[4];
    acc[0] = vzero; acc[1] = vzero; acc[2] = vzero; acc[3] = vzero;

    const half_t* arow = A + (size_t)(m0 + r15) * lda + hi * 8;
    const half_t* brow0 = Bt + (size_t)(n0 + r15) * K + hi * 8;

    for (int kb = 0; kb < K; kb += 32) {
        v8h a0 = *(const v8h*)(arow + kb);
        v8h a1 = *(const v8h*)(arow + kb + 16);
        v16h af;
        #pragma unroll
        for (int i = 0; i < 8; ++i) { af[i] = a0[i]; af[i + 8] = a1[i]; }

        #pragma unroll
        for (int tt = 0; tt < 4; ++tt) {
            const half_t* brow = brow0 + (size_t)(tt * 16) * K + kb;
            v8h b0 = *(const v8h*)brow;
            v8h b1 = *(const v8h*)(brow + 16);
            v16h bf;
            #pragma unroll
            for (int i = 0; i < 8; ++i) { bf[i] = b0[i]; bf[i + 8] = b1[i]; }
            acc[tt] = __builtin_amdgcn_wmma_f32_16x16x32_f16(
                false, af, false, bf, (short)0, acc[tt], false, false);
        }
    }

    #pragma unroll
    for (int tt = 0; tt < 4; ++tt) {
        const int n  = n0 + tt * 16 + r15;
        const float bv = bias ? bias[n] : 0.f;
        #pragma unroll
        for (int r = 0; r < 8; ++r) {
            const int m = m0 + hi * 8 + r;    // ISA C layout: vgpr r, hi half -> M=8+r
            float v = acc[tt][r] + bv;
            if (act == 1)      v = fmaxf(v, 0.f);
            else if (act == 2) v = sigf(v);
            C[(size_t)m * ldc + n] = v;
            if (C16) C16[(size_t)m * ldc16 + n] = (half_t)v;
        }
    }
}

// ---------------------------------------------------------------------------
// GRU cell: gates (r,z,n); h updated in place (fp32), fp16 mirror written
// ---------------------------------------------------------------------------
__global__ void gru_cell_kernel(const float* __restrict__ gi,
                                const float* __restrict__ gh,
                                float* __restrict__ h,
                                half_t* __restrict__ h16, int ld16,
                                int M, int H)
{
    int id = blockIdx.x * blockDim.x + threadIdx.x;
    if (id >= M * H) return;
    int row = id / H, i = id % H;
    const float* gir = gi + (size_t)row * 3 * H;
    const float* ghr = gh + (size_t)row * 3 * H;
    float r = sigf(gir[i]         + ghr[i]);
    float z = sigf(gir[H + i]     + ghr[H + i]);
    float n = tanhf(gir[2*H + i] + r * ghr[2*H + i]);
    float hp = h[(size_t)row * H + i];
    float hn = (1.f - z) * n + z * hp;
    h[(size_t)row * H + i] = hn;
    h16[(size_t)row * ld16 + i] = (half_t)hn;
}

// W[K,N] f32 -> Wt[N,K] f16
__global__ void wconvT_kernel(const float* __restrict__ W, half_t* __restrict__ Wt,
                              int K, int N)
{
    int id = blockIdx.x * blockDim.x + threadIdx.x;
    if (id >= K * N) return;
    int n = id / K, k = id % K;
    Wt[(size_t)n * K + k] = (half_t)W[(size_t)k * N + n];
}

__global__ void zero_f32_kernel(float* p, int n) {
    int id = blockIdx.x * blockDim.x + threadIdx.x;
    if (id < n) p[id] = 0.f;
}
__global__ void zero_f16_kernel(half_t* p, int n) {
    int id = blockIdx.x * blockDim.x + threadIdx.x;
    if (id < n) p[id] = (half_t)0.f;
}
// pb[0:n) = b, pb[n:ntot) = 0
__global__ void padbias_kernel(const float* __restrict__ b, float* __restrict__ pb,
                               int n, int ntot) {
    int id = blockIdx.x * blockDim.x + threadIdx.x;
    if (id < ntot) pb[id] = (id < n) ? b[id] : 0.f;
}

// e = relu(x @ enc_emb_w + b), swapaxes -> seq0[T,B,32] f16.  x:[B,T,4]
__global__ void enc_emb_kernel(const float* __restrict__ x, const float* __restrict__ W,
                               const float* __restrict__ b, half_t* __restrict__ seq0)
{
    int id = blockIdx.x * blockDim.x + threadIdx.x;
    if (id >= TT * BB * 32) return;
    int j = id % 32, tb = id / 32;
    int bb = tb % BB, t = tb / BB;
    float a = b[j];
    #pragma unroll
    for (int i = 0; i < 4; ++i) a += x[((size_t)bb * TT + t) * 4 + i] * W[i * 32 + j];
    seq0[id] = (half_t)fmaxf(a, 0.f);
}

// enc_x head: m2[65536,64(padded, first 32 valid)] -> [65536,4] sigmoid
__global__ void ep3_kernel(const float* __restrict__ m2, const float* __restrict__ W,
                           const float* __restrict__ b, float* __restrict__ encx)
{
    int id = blockIdx.x * blockDim.x + threadIdx.x;
    if (id >= TT * BB * 4) return;
    int o = id % 4; size_t row = id / 4;
    float a = b[o];
    for (int k = 0; k < 32; ++k) a += m2[row * 64 + k] * W[k * 4 + o];
    encx[id] = sigf(a);
}

// x_emb into dec_in[:, :, 0:128]
__global__ void xemb_kernel(const float* __restrict__ encx, const float* __restrict__ W,
                            const float* __restrict__ b, half_t* __restrict__ dec_in)
{
    int id = blockIdx.x * blockDim.x + threadIdx.x;
    if (id >= TT * BB * 128) return;
    int j = id % 128; size_t tb = id / 128;
    float a = b[j];
    #pragma unroll
    for (int i = 0; i < 4; ++i) a += encx[tb * 4 + i] * W[i * 128 + j];
    dec_in[tb * 256 + j] = (half_t)fmaxf(a, 0.f);
}

// tokens t>=1: relu(y[:,t-1] @ yemb) into dec_in[t,:,128:256].  y:[B,T,9]
__global__ void ytok_kernel(const float* __restrict__ y, const float* __restrict__ W,
                            const float* __restrict__ b, half_t* __restrict__ dec_in)
{
    int id = blockIdx.x * blockDim.x + threadIdx.x;
    if (id >= (TT - 1) * BB * 128) return;
    int j = id % 128; int rb = id / 128;
    int bb = rb % BB, tm1 = rb / BB;           // token index t = tm1+1
    float a = b[j];
    #pragma unroll
    for (int i = 0; i < 9; ++i) a += y[((size_t)bb * TT + tm1) * 9 + i] * W[i * 128 + j];
    dec_in[((size_t)(tm1 + 1) * BB + bb) * 256 + 128 + j] = (half_t)fmaxf(a, 0.f);
}

// token t=0: raw start embedding lookup
__global__ void tok0_kernel(const int* __restrict__ sidx, const float* __restrict__ emb,
                            half_t* __restrict__ dec_in)
{
    int id = blockIdx.x * blockDim.x + threadIdx.x;
    if (id >= BB * 128) return;
    int j = id % 128, bb = id / 128;
    dec_in[(size_t)bb * 256 + 128 + j] = (half_t)emb[(size_t)sidx[bb] * 128 + j];
}

// split relu(stl) output [1024,3072] -> three h states (f32 + f16 mirrors)
__global__ void split_h_kernel(const float* __restrict__ hinit,
                               float* h0, float* h1, float* h2,
                               half_t* g0, half_t* g1, half_t* g2)
{
    int id = blockIdx.x * blockDim.x + threadIdx.x;
    if (id >= BB * 3072) return;
    int bb = id / 3072, j = id % 3072;
    int l = j >> 10, i = j & 1023;
    float v = hinit[id];
    float*  hf = (l == 0) ? h0 : (l == 1) ? h1 : h2;
    half_t* hh = (l == 0) ? g0 : (l == 1) ? g1 : g2;
    hf[(size_t)bb * 1024 + i] = v;
    hh[(size_t)bb * 1024 + i] = (half_t)v;
}

// vel head per timestep: [1024,64] -> sigmoid 9, write vel[b,o,t]
__global__ void dp3_kernel(const float* __restrict__ cF, const float* __restrict__ W,
                           const float* __restrict__ b, float* __restrict__ vel, int t)
{
    int id = blockIdx.x * blockDim.x + threadIdx.x;
    if (id >= BB * 9) return;
    int o = id % 9, bb = id / 9;
    float a = b[o];
    for (int k = 0; k < 64; ++k) a += cF[(size_t)bb * 64 + k] * W[k * 9 + o];
    vel[(size_t)bb * 576 + o * 64 + t] = sigf(a);
}

// conv1d SAME k=3:  vel[B,9,64] -> relu -> c1[B,32,64]
__global__ void conv1_kernel(const float* __restrict__ vel, const float* __restrict__ W,
                             const float* __restrict__ b, float* __restrict__ c1)
{
    int id = blockIdx.x * blockDim.x + threadIdx.x;
    if (id >= BB * 32 * 64) return;
    int t = id % 64, rest = id / 64;
    int o = rest % 32, bb = rest / 32;
    float a = b[o];
    for (int i = 0; i < 9; ++i)
        #pragma unroll
        for (int k = 0; k < 3; ++k) {
            int ts = t + k - 1;
            if (ts >= 0 && ts < 64)
                a += vel[(size_t)bb * 576 + i * 64 + ts] * W[(o * 9 + i) * 3 + k];
        }
    c1[id] = fmaxf(a, 0.f);
}

// conv1d SAME k=3:  c1[B,32,64] -> sigmoid -> onset[B,9,64]
__global__ void conv2_kernel(const float* __restrict__ c1, const float* __restrict__ W,
                             const float* __restrict__ b, float* __restrict__ onset)
{
    int id = blockIdx.x * blockDim.x + threadIdx.x;
    if (id >= BB * 9 * 64) return;
    int t = id % 64, rest = id / 64;
    int o = rest % 9, bb = rest / 9;
    float a = b[o];
    for (int i = 0; i < 32; ++i)
        #pragma unroll
        for (int k = 0; k < 3; ++k) {
            int ts = t + k - 1;
            if (ts >= 0 && ts < 64)
                a += c1[((size_t)bb * 32 + i) * 64 + ts] * W[(o * 32 + i) * 3 + k];
        }
    onset[(size_t)bb * 576 + o * 64 + t] = sigf(a);
}

// ---------------------------------------------------------------------------
extern "C" void kernel_launch(void* const* d_in, const int* in_sizes, int n_in,
                              void* d_out, int out_size, void* d_ws, size_t ws_size,
                              hipStream_t stream)
{
    const float* x_in  = (const float*)d_in[0];
    const int*   sidx  = (const int*)  d_in[1];
    const float* y_in  = (const float*)d_in[2];
    #define PF(i) ((const float*)d_in[3 + (i)])
    // param indices (dict insertion order):
    // 0 enc_emb_w, 1 enc_emb_b; enc GRU: base 2 + l*8 + dir*4 -> Wi,Wh,bi,bh
    // 26..33 ep0..3 (w,b); 34 xemb_w 35 xemb_b; 36 yemb_w 37 yemb_b; 38 start_emb
    // 39 stl_w 40 stl_b; dec GRU: 41 + l*4 -> Wi,Wh,bi,bh; 53..60 dp0..3 (w,b)
    // 61 ow1 62 ob1 63 ow2 64 ob2

    // ---- workspace bump allocator ----
    size_t off = 0;
    auto alloc = [&](size_t bytes) -> void* {
        size_t o = (off + 255) & ~(size_t)255;
        off = o + bytes;
        return (void*)((char*)d_ws + o);
    };
    auto aH = [&](size_t n) { return (half_t*)alloc(n * sizeof(half_t)); };
    auto aF = [&](size_t n) { return (float*) alloc(n * sizeof(float)); };

    // fp16 transposed weights
    half_t* egWiT[3][2]; half_t* egWhT[3][2];
    for (int l = 0; l < 3; ++l)
        for (int d = 0; d < 2; ++d) {
            int din = (l == 0) ? 32 : 256;
            egWiT[l][d] = aH((size_t)384 * din);
            egWhT[l][d] = aH((size_t)384 * 128);
        }
    half_t* ep0T = aH(128 * 256);
    half_t* ep1T = aH(64 * 128);
    half_t* ep2T = aH(64 * 64);          // padded: N 32 -> 64
    float*  ep2bP = aF(64);              // padded bias
    half_t* stlT = aH((size_t)3072 * 128);
    half_t* dgWiT[3]; half_t* dgWhT[3];
    for (int l = 0; l < 3; ++l) {
        int din = (l == 0) ? 256 : 1024;
        dgWiT[l] = aH((size_t)3072 * din);
        dgWhT[l] = aH((size_t)3072 * 1024);
    }
    half_t* dp0T = aH((size_t)512 * 1024);
    half_t* dp1T = aH((size_t)256 * 512);
    half_t* dp2T = aH((size_t)64 * 256);

    // activations
    half_t* seq0   = aH((size_t)TT * BB * 32);
    half_t* seqA   = aH((size_t)TT * BB * 256);
    half_t* seqB   = aH((size_t)TT * BB * 256);
    half_t* dec_in = aH((size_t)TT * BB * 256);
    float*  giB    = aF((size_t)BB * 3072);
    float*  ghB    = aF((size_t)BB * 3072);
    float*  henc   = aF((size_t)BB * 128);
    half_t* zero128= aH((size_t)BB * 128);
    float*  hinit  = aF((size_t)BB * 3072);
    float*  hd32[3]; half_t* hd16[3];
    for (int l = 0; l < 3; ++l) { hd32[l] = aF((size_t)BB * 1024); hd16[l] = aH((size_t)BB * 1024); }
    float*  m0  = aF((size_t)TT * BB * 128); half_t* m0h = aH((size_t)TT * BB * 128);
    float*  m1  = aF((size_t)TT * BB * 64);  half_t* m1h = aH((size_t)TT * BB * 64);
    float*  m2  = aF((size_t)TT * BB * 64);  // padded cols (first 32 valid)
    float*  encx= aF((size_t)TT * BB * 4);
    float*  mA  = aF((size_t)BB * 512); half_t* mAh = aH((size_t)BB * 512);
    float*  mB  = aF((size_t)BB * 256); half_t* mBh = aH((size_t)BB * 256);
    float*  mC  = aF((size_t)BB * 64);
    float*  c1  = aF((size_t)BB * 32 * 64);
    (void)ws_size; (void)in_sizes; (void)n_in; (void)out_size;

    float* vel   = (float*)d_out;
    float* onset = (float*)d_out + (size_t)BB * 9 * TT;

    // Exact-cover GEMM launch: N%64==0, strips = N/64, waves/block <= 8 and
    // dividing the strip count so no bounds checks are needed in-kernel.
    auto gemm = [&](const half_t* A, int lda, const half_t* Bt,
                    float* C, int ldc, half_t* C16, int ldc16,
                    const float* bias, int M, int N, int K, int act) {
        int strips = N / 64;
        int wpb = strips < 8 ? strips : 8;     // 1,2,4,6,8 all divide our strip counts
        dim3 g((unsigned)(strips / wpb), (unsigned)(M / 16));
        gemm16_kernel<<<g, wpb * 32, 0, stream>>>(A, lda, Bt, C, ldc, C16, ldc16,
                                                  bias, K, act);
    };
    auto wconvT = [&](const float* W, half_t* Wt, int K, int N) {
        int n = K * N;
        wconvT_kernel<<<(n + 255) / 256, 256, 0, stream>>>(W, Wt, K, N);
    };
    auto cell = [&](const float* gi, const float* gh, float* h, half_t* h16, int ld16,
                    int M, int H) {
        int n = M * H;
        gru_cell_kernel<<<(n + 255) / 256, 256, 0, stream>>>(gi, gh, h, h16, ld16, M, H);
    };

    // ---- 1. weight conversion ----
    for (int l = 0; l < 3; ++l)
        for (int d = 0; d < 2; ++d) {
            int base = 2 + l * 8 + d * 4;
            int din  = (l == 0) ? 32 : 256;
            wconvT(PF(base + 0), egWiT[l][d], din, 384);
            wconvT(PF(base + 1), egWhT[l][d], 128, 384);
        }
    wconvT(PF(26), ep0T, 256, 128);
    wconvT(PF(28), ep1T, 128, 64);
    zero_f16_kernel<<<(64 * 64 + 255) / 256, 256, 0, stream>>>(ep2T, 64 * 64);
    wconvT(PF(30), ep2T, 64, 32);                       // fills rows 0..31 of 64
    padbias_kernel<<<1, 64, 0, stream>>>(PF(31), ep2bP, 32, 64);
    wconvT(PF(39), stlT, 128, 3072);
    for (int l = 0; l < 3; ++l) {
        int din = (l == 0) ? 256 : 1024;
        wconvT(PF(41 + l * 4), dgWiT[l], din, 3072);
        wconvT(PF(42 + l * 4), dgWhT[l], 1024, 3072);
    }
    wconvT(PF(53), dp0T, 1024, 512);
    wconvT(PF(55), dp1T, 512, 256);
    wconvT(PF(57), dp2T, 256, 64);

    // ---- 2. encoder embedding + zero buffers ----
    {
        int n = TT * BB * 32;
        enc_emb_kernel<<<(n + 255) / 256, 256, 0, stream>>>(x_in, PF(0), PF(1), seq0);
        int z = BB * 128;
        zero_f16_kernel<<<(z + 255) / 256, 256, 0, stream>>>(zero128, z);
    }

    // ---- 3. 3-layer bidirectional encoder GRU ----
    for (int l = 0; l < 3; ++l) {
        const half_t* seq_in = (l == 0) ? seq0 : (l == 1) ? seqA : seqB;
        int ldin             = (l == 0) ? 32 : 256;
        half_t* seq_out      = (l == 0) ? seqA : (l == 1) ? seqB : seqA;
        int din              = (l == 0) ? 32 : 256;
        for (int d = 0; d < 2; ++d) {
            int base = 2 + l * 8 + d * 4;
            const float* bi = PF(base + 2);
            const float* bh = PF(base + 3);
            int z = BB * 128;
            zero_f32_kernel<<<(z + 255) / 256, 256, 0, stream>>>(henc, z);
            for (int s = 0; s < TT; ++s) {
                int t = (d == 0) ? s : (TT - 1 - s);
                gemm(seq_in + (size_t)t * BB * ldin, ldin, egWiT[l][d],
                     giB, 384, nullptr, 0, bi, BB, 384, din, 0);
                const half_t* hA; int hld;
                if (s == 0) { hA = zero128; hld = 128; }
                else {
                    int tp = (d == 0) ? t - 1 : t + 1;
                    hA = seq_out + (size_t)tp * BB * 256 + d * 128; hld = 256;
                }
                gemm(hA, hld, egWhT[l][d], ghB, 384, nullptr, 0, bh, BB, 384, 128, 0);
                cell(giB, ghB, henc, seq_out + (size_t)t * BB * 256 + d * 128, 256, BB, 128);
            }
        }
    }

    // ---- 4. encoder projection MLP (batched over T*B) ----
    gemm(seqA, 256, ep0T, m0, 128, m0h, 128, PF(27), TT * BB, 128, 256, 1);
    gemm(m0h, 128, ep1T, m1, 64,  m1h, 64,  PF(29), TT * BB, 64, 128, 1);
    gemm(m1h, 64,  ep2T, m2, 64,  nullptr, 0, ep2bP, TT * BB, 64, 64, 1);  // padded N
    {
        int n = TT * BB * 4;
        ep3_kernel<<<(n + 255) / 256, 256, 0, stream>>>(m2, PF(32), PF(33), encx);
    }

    // ---- 5. decoder inputs ----
    {
        int n = TT * BB * 128;
        xemb_kernel<<<(n + 255) / 256, 256, 0, stream>>>(encx, PF(34), PF(35), dec_in);
        n = (TT - 1) * BB * 128;
        ytok_kernel<<<(n + 255) / 256, 256, 0, stream>>>(y_in, PF(36), PF(37), dec_in);
        n = BB * 128;
        tok0_kernel<<<(n + 255) / 256, 256, 0, stream>>>(sidx, PF(38), dec_in);
    }

    // ---- 6. initial decoder hidden: relu(tok0 @ stl_w + b) -> split ----
    gemm(dec_in + 128, 256, stlT, hinit, 3072, nullptr, 0, PF(40), BB, 3072, 128, 1);
    {
        int n = BB * 3072;
        split_h_kernel<<<(n + 255) / 256, 256, 0, stream>>>(
            hinit, hd32[0], hd32[1], hd32[2], hd16[0], hd16[1], hd16[2]);
    }

    // ---- 7. decoder: 3 stacked GRU layers fused per timestep + MLP head ----
    for (int t = 0; t < TT; ++t) {
        // layer 0: input dec_in[t] (K=256)
        gemm(dec_in + (size_t)t * BB * 256, 256, dgWiT[0], giB, 3072, nullptr, 0,
             PF(43), BB, 3072, 256, 0);
        gemm(hd16[0], 1024, dgWhT[0], ghB, 3072, nullptr, 0, PF(44), BB, 3072, 1024, 0);
        cell(giB, ghB, hd32[0], hd16[0], 1024, BB, 1024);
        // layer 1: input = layer0 hidden
        gemm(hd16[0], 1024, dgWiT[1], giB, 3072, nullptr, 0, PF(47), BB, 3072, 1024, 0);
        gemm(hd16[1], 1024, dgWhT[1], ghB, 3072, nullptr, 0, PF(48), BB, 3072, 1024, 0);
        cell(giB, ghB, hd32[1], hd16[1], 1024, BB, 1024);
        // layer 2
        gemm(hd16[1], 1024, dgWiT[2], giB, 3072, nullptr, 0, PF(51), BB, 3072, 1024, 0);
        gemm(hd16[2], 1024, dgWhT[2], ghB, 3072, nullptr, 0, PF(52), BB, 3072, 1024, 0);
        cell(giB, ghB, hd32[2], hd16[2], 1024, BB, 1024);
        // projection MLP 1024->512->256->64->9
        gemm(hd16[2], 1024, dp0T, mA, 512, mAh, 512, PF(54), BB, 512, 1024, 1);
        gemm(mAh, 512, dp1T, mB, 256, mBh, 256, PF(56), BB, 256, 512, 1);
        gemm(mBh, 256, dp2T, mC, 64, nullptr, 0, PF(58), BB, 64, 256, 1);
        int n = BB * 9;
        dp3_kernel<<<(n + 255) / 256, 256, 0, stream>>>(mC, PF(59), PF(60), vel, t);
    }

    // ---- 8. onset CNN ----
    {
        int n = BB * 32 * 64;
        conv1_kernel<<<(n + 255) / 256, 256, 0, stream>>>(vel, PF(61), PF(62), c1);
        n = BB * 9 * 64;
        conv2_kernel<<<(n + 255) / 256, 256, 0, stream>>>(c1, PF(63), PF(64), onset);
    }
    #undef PF
}
